// SelfAttention2d_1898375545162
// MI455X (gfx1250) — matrix operand order repaired and checked by reading, not compiled, not run on hardware
//
#include <hip/hip_runtime.h>
#include <hip/hip_bf16.h>

// ---------------------------------------------------------------------------
// SelfAttention2d for MI455X (gfx1250): AdaLN + MHSA
// v_wmma_f32_16x16x32_f16 matrix cores + GLOBAL_LOAD_ASYNC_TO_LDS staging
// ---------------------------------------------------------------------------

typedef __attribute__((ext_vector_type(16))) _Float16 v16h;
typedef __attribute__((ext_vector_type(8)))  float    v8f;
typedef int i4 __attribute__((vector_size(16)));
typedef __attribute__((address_space(1))) i4* gas_i4p;
typedef __attribute__((address_space(3))) i4* las_i4p;

#define EPSV 1e-5f
#define NB   8      // batch
#define CC   512    // channels
#define SS   4096   // h*w
#define NHD  8      // heads
#define DHD  64     // head dim

#if __has_builtin(__builtin_amdgcn_global_load_async_to_lds_b128) && \
    __has_builtin(__builtin_amdgcn_s_wait_asynccnt)
#define USE_ASYNC 1
#else
#define USE_ASYNC 0
#endif

// 16-byte tile-staging copy: async DMA to LDS on gfx1250, else manual b128.
static __device__ __forceinline__ void cp16(_Float16* lds_dst, const _Float16* gsrc) {
#if USE_ASYNC
  __builtin_amdgcn_global_load_async_to_lds_b128(
      (gas_i4p)gsrc, (las_i4p)lds_dst, /*offset=*/0, /*cpol=*/0);
#else
  *(i4*)lds_dst = *(const i4*)gsrc;
#endif
}

static __device__ __forceinline__ void async_wait() {
#if USE_ASYNC
  __builtin_amdgcn_s_wait_asynccnt(0);
#endif
}

static __device__ __forceinline__ v8f wmma16(v16h a, v16h b, v8f c) {
  // D = A(16x32 f16) * B(32x16 f16) + C(16x16 f32)
  return __builtin_amdgcn_wmma_f32_16x16x32_f16(
      /*neg_a=*/false, a, /*neg_b=*/false, b,
      /*c_mod=*/(short)0, c, /*reuse_a=*/false, /*reuse_b=*/false);
}

// A fragment: matrix element (M,K) at mem[M*stride + K]  (stride even)
// lane L: M = L&15, half = L>>4 ; VGPR i: K = (i<4?0:16) + 8*half + 2*(i&3)
static __device__ __forceinline__ v16h frag_A(const _Float16* p, int stride, int lane) {
  union { v16h h; unsigned int u[8]; } f;
  const unsigned int* p32 = (const unsigned int*)p;
  int M = lane & 15, half = lane >> 4;
  int s2 = stride >> 1;
#pragma unroll
  for (int i = 0; i < 8; ++i) {
    int K = ((i & 4) << 2) + (half << 3) + ((i & 3) << 1);
    f.u[i] = p32[M * s2 + (K >> 1)];
  }
  return f.h;
}

// B fragment: matrix element (K,N) at mem[N*stride + K]
// lane L: N = L&15, half = L>>4 ; VGPR v: K = (v<4?0:16) + 8*half + 2*(v&3)
static __device__ __forceinline__ v16h frag_B(const _Float16* p, int stride, int lane) {
  union { v16h h; unsigned int u[8]; } f;
  const unsigned int* p32 = (const unsigned int*)p;
  int N = lane & 15, half = lane >> 4;
  int s2 = stride >> 1;
#pragma unroll
  for (int v = 0; v < 8; ++v) {
    f.u[v] = p32[N * s2 + ((v & 4) << 1) + (half << 2) + (v & 3)];
  }
  return f.h;
}

// ---------------------------------------------------------------------------
// 0) pre-convert W_qkv (1536x512) and W_out (512x512) to f16 (once)
// ---------------------------------------------------------------------------
__global__ __launch_bounds__(256) void sa2d_cvtw(const float* __restrict__ Wq,
                                                 const float* __restrict__ Wo,
                                                 _Float16* __restrict__ wq16,
                                                 _Float16* __restrict__ wo16) {
  int base = blockIdx.x * 1024 + threadIdx.x;
#pragma unroll
  for (int j = 0; j < 4; ++j) {
    int e = base + (j << 8);
    if (e < 1536 * 512) wq16[e] = (_Float16)Wq[e];
    else                wo16[e - 1536 * 512] = (_Float16)Wo[e - 1536 * 512];
  }
}

// ---------------------------------------------------------------------------
// 1) scales_shifts = cond @ W_ada + b_ada   (8 x 1024, tiny)
// ---------------------------------------------------------------------------
__global__ __launch_bounds__(256) void sa2d_ada(const float* __restrict__ cond,
                                                const float* __restrict__ W_ada,
                                                const float* __restrict__ b_ada,
                                                float* __restrict__ ss) {
  int n = blockIdx.x, tid = threadIdx.x;
  const float* cn = cond + n * 512;
#pragma unroll
  for (int j = 0; j < 4; ++j) {
    int o = tid + (j << 8);
    float acc = b_ada[o];
    for (int k = 0; k < 512; ++k) acc += cn[k] * W_ada[k * 1024 + o];
    ss[n * 1024 + o] = acc;
  }
}

// ---------------------------------------------------------------------------
// 2) per-batch mean / rstd over 2M elements (deterministic tree reduction)
// ---------------------------------------------------------------------------
__global__ __launch_bounds__(256) void sa2d_red1(const float* __restrict__ x,
                                                 float* __restrict__ partials) {
  int n = blockIdx.y, bx = blockIdx.x, tid = threadIdx.x;
  const float* xb = x + (size_t)n * (CC * SS);
  int idx = bx * 256 + tid;
  float s = 0.f, q = 0.f;
  for (int i = 0; i < 128; ++i) { float v = xb[idx + i * 16384]; s += v; q += v * v; }
  __shared__ float s1[256], s2[256];
  s1[tid] = s; s2[tid] = q; __syncthreads();
  for (int off = 128; off > 0; off >>= 1) {
    if (tid < off) { s1[tid] += s1[tid + off]; s2[tid] += s2[tid + off]; }
    __syncthreads();
  }
  if (tid == 0) { partials[(n * 64 + bx) * 2] = s1[0]; partials[(n * 64 + bx) * 2 + 1] = s2[0]; }
}

__global__ __launch_bounds__(64) void sa2d_red2(const float* __restrict__ partials,
                                                float* __restrict__ stats) {
  int n = blockIdx.x, tid = threadIdx.x;
  __shared__ float s1[64], s2[64];
  s1[tid] = partials[(n * 64 + tid) * 2];
  s2[tid] = partials[(n * 64 + tid) * 2 + 1];
  __syncthreads();
  for (int off = 32; off > 0; off >>= 1) {
    if (tid < off) { s1[tid] += s1[tid + off]; s2[tid] += s2[tid + off]; }
    __syncthreads();
  }
  if (tid == 0) {
    float cnt = (float)(CC * SS);
    float mean = s1[0] / cnt;
    float var = s2[0] / cnt - mean * mean;
    stats[n * 2] = mean;
    stats[n * 2 + 1] = rsqrtf(var + EPSV);
  }
}

// ---------------------------------------------------------------------------
// 3) fused AdaLN + QKV projection: qkv = W_qkv @ norm(x)  (1536x512 @ 512x4096)
//    qkvbuf[n][g][s][d] f16; g: q=0..7, k=8..15, v=16..23
//    LDS tiles stride 40 halves (80B rows: 16B-aligned for async, 20-bank stride
//    -> conflict-free fragment reads)
// ---------------------------------------------------------------------------
__global__ __launch_bounds__(128) void sa2d_qkv(const float* __restrict__ x,
                                                const _Float16* __restrict__ wq16,
                                                const float* __restrict__ b_qkv,
                                                const float* __restrict__ ss,
                                                const float* __restrict__ stats,
                                                _Float16* __restrict__ qkvbuf) {
  int n = blockIdx.z, ob = blockIdx.y * 64, sb = blockIdx.x * 64;
  int tid = threadIdx.x, lane = tid & 31, wv = tid >> 5;
  __shared__ __align__(16) _Float16 aT[64 * 40];  // [o][c]  W tile
  __shared__ __align__(16) _Float16 bT[64 * 40];  // [s][c]  normed x tile
  float mean = stats[n * 2], rstd = stats[n * 2 + 1];
  v8f cacc[2][2] = {};
  int mrow = (wv >> 1) << 5, ncol = (wv & 1) << 5;

  for (int kc = 0; kc < 512; kc += 32) {
    // W tile 64x32 f16: 256 16B-chunks, async DMA to LDS
#pragma unroll
    for (int i = 0; i < 2; ++i) {
      int lin = tid + (i << 7);
      int oI = lin >> 2, ch = lin & 3;
      cp16(aT + oI * 40 + ch * 8, wq16 + (size_t)(ob + oI) * 512 + kc + ch * 8);
    }
    // normed x tile 32(c) x 64(s): computed, s-fast for coalescing
#pragma unroll
    for (int i = 0; i < 16; ++i) {
      int lin = tid + (i << 7);
      int sI = lin & 63, cI = lin >> 6;
      int chn = kc + cI;
      float xv = x[((size_t)n * CC + chn) * SS + sb + sI];
      float sc = ss[n * 1024 + chn] + 1.0f;
      float sh = ss[n * 1024 + 512 + chn];
      bT[sI * 40 + cI] = (_Float16)(sh + (xv - mean) * rstd * sc);
    }
    async_wait();
    __syncthreads();
    v16h A0 = frag_A(aT + mrow * 40, 40, lane);
    v16h A1 = frag_A(aT + (mrow + 16) * 40, 40, lane);
    v16h B0 = frag_B(bT + ncol * 40, 40, lane);
    v16h B1 = frag_B(bT + (ncol + 16) * 40, 40, lane);
    cacc[0][0] = wmma16(A0, B0, cacc[0][0]);
    cacc[0][1] = wmma16(A0, B1, cacc[0][1]);
    cacc[1][0] = wmma16(A1, B0, cacc[1][0]);
    cacc[1][1] = wmma16(A1, B1, cacc[1][1]);
    __syncthreads();
  }

  int half = lane >> 4, nn = lane & 15;
#pragma unroll
  for (int mi = 0; mi < 2; ++mi)
#pragma unroll
    for (int ni = 0; ni < 2; ++ni)
#pragma unroll
      for (int v = 0; v < 8; ++v) {
        int o = ob + mrow + (mi << 4) + v + (half << 3);
        int sIdx = sb + ncol + (ni << 4) + nn;
        float val = cacc[mi][ni][v] + b_qkv[o];
        int g = o >> 6, d = o & 63;
        qkvbuf[(((size_t)n * 24 + g) * SS + sIdx) * DHD + d] = (_Float16)val;
      }
}

// ---------------------------------------------------------------------------
// 4) Flash attention per (n, head): 128 query rows/block, 64-key tiles,
//    online softmax; K tile async-DMA'd to LDS (stride 72 -> conflict-free),
//    V staged transposed, P converted C->A layout via per-wave LDS (stride 72).
// ---------------------------------------------------------------------------
__global__ __launch_bounds__(256) void sa2d_flash(const _Float16* __restrict__ qkvbuf,
                                                  _Float16* __restrict__ ybuf) {
  int nh = blockIdx.y;
  int n = nh >> 3, head = nh & 7;
  int qb = blockIdx.x * 128;
  int tid = threadIdx.x, lane = tid & 31, wv = tid >> 5;

  __shared__ __align__(16) _Float16 kT[64 * 72];      // [t][d]  keys
  __shared__ __align__(16) _Float16 vT[64 * 66];      // [d][t]  values (transposed)
  __shared__ __align__(16) _Float16 pS[8 * 16 * 72];  // per-wave P staging

  const _Float16* qptr =
      qkvbuf + (((size_t)n * 24 + head) * SS + qb + wv * 16) * DHD;
  v16h QA0 = frag_A(qptr, 64, lane);
  v16h QA1 = frag_A(qptr + 32, 64, lane);

  const _Float16* khead = qkvbuf + (((size_t)n * 24 + 8 + head) * SS) * DHD;
  const unsigned int* vsrc =
      (const unsigned int*)(qkvbuf + (((size_t)n * 24 + 16 + head) * SS) * DHD);

  float m[8], l[8];
  v8f acc[4] = {};
#pragma unroll
  for (int v = 0; v < 8; ++v) { m[v] = -1e30f; l[v] = 0.f; }

  int half = lane >> 4;
  _Float16* pw = pS + wv * 16 * 72;

  for (int kt = 0; kt < 64; ++kt) {
    int tb = kt * 64;
    __syncthreads();
    // K tile 64x64 f16 = 512 16B chunks: async DMA into padded LDS
#pragma unroll
    for (int i = 0; i < 2; ++i) {
      int lin = tid + (i << 8);
      int t = lin >> 3, ch = lin & 7;
      cp16(kT + t * 72 + ch * 8, khead + (size_t)(tb + t) * 64 + ch * 8);
    }
    // V tile transposed: [d][t], padded stride 66
#pragma unroll
    for (int i = 0; i < 8; ++i) {
      int lin = tid + (i << 8);
      int t = lin >> 5, dp = lin & 31;
      unsigned int wvv = vsrc[(size_t)(tb + t) * 32 + dp];
      union { unsigned int u; _Float16 h[2]; } cvt; cvt.u = wvv;
      vT[(2 * dp) * 66 + t]     = cvt.h[0];
      vT[(2 * dp + 1) * 66 + t] = cvt.h[1];
    }
    async_wait();
    __syncthreads();

    // S = Q @ K^T  (16 queries x 64 keys), scale 1/sqrt(dh) = 0.125
    v8f sc[4] = {};
#pragma unroll
    for (int j = 0; j < 4; ++j) {
      v16h B0 = frag_B(kT + (j * 16) * 72, 72, lane);       // d = 0..31
      v16h B1 = frag_B(kT + (j * 16) * 72 + 32, 72, lane);  // d = 32..63
      sc[j] = wmma16(QA0, B0, sc[j]);
      sc[j] = wmma16(QA1, B1, sc[j]);
    }
#pragma unroll
    for (int j = 0; j < 4; ++j)
#pragma unroll
      for (int v = 0; v < 8; ++v) sc[j][v] *= 0.125f;

    // online softmax: row stats per (VGPR v, lane half) in 16-lane groups
    float alpha[8];
#pragma unroll
    for (int v = 0; v < 8; ++v) {
      float mx = fmaxf(fmaxf(sc[0][v], sc[1][v]), fmaxf(sc[2][v], sc[3][v]));
#pragma unroll
      for (int off = 1; off < 16; off <<= 1) mx = fmaxf(mx, __shfl_xor(mx, off, 32));
      float mn = fmaxf(m[v], mx);
      alpha[v] = __expf(m[v] - mn);
      m[v] = mn;
      float rs = 0.f;
#pragma unroll
      for (int j = 0; j < 4; ++j) { float p = __expf(sc[j][v] - mn); sc[j][v] = p; rs += p; }
#pragma unroll
      for (int off = 1; off < 16; off <<= 1) rs += __shfl_xor(rs, off, 32);
      l[v] = l[v] * alpha[v] + rs;
    }
#pragma unroll
    for (int j = 0; j < 4; ++j)
#pragma unroll
      for (int v = 0; v < 8; ++v) acc[j][v] *= alpha[v];

    // P: C layout -> A layout via per-wave LDS
#pragma unroll
    for (int j = 0; j < 4; ++j)
#pragma unroll
      for (int v = 0; v < 8; ++v)
        pw[(v + (half << 3)) * 72 + (j << 4) + (lane & 15)] = (_Float16)sc[j][v];
    __syncthreads();

    v16h PA0 = frag_A(pw, 72, lane);
    v16h PA1 = frag_A(pw + 32, 72, lane);
#pragma unroll
    for (int j = 0; j < 4; ++j) {  // acc += P @ V (dh tile j)
      v16h VB0 = frag_B(vT + (j * 16) * 66, 66, lane);       // t = 0..31
      v16h VB1 = frag_B(vT + (j * 16) * 66 + 32, 66, lane);  // t = 32..63
      acc[j] = wmma16(PA0, VB0, acc[j]);
      acc[j] = wmma16(PA1, VB1, acc[j]);
    }
  }

  // epilogue: y = acc / l -> ybuf[n][head][s][d] f16
  _Float16* yb = ybuf + (((size_t)n * NHD + head) * SS + qb + wv * 16) * DHD;
  int nn = lane & 15;
#pragma unroll
  for (int v = 0; v < 8; ++v) {
    float inv = 1.0f / l[v];
    int row = v + (half << 3);
#pragma unroll
    for (int j = 0; j < 4; ++j)
      yb[row * 64 + (j << 4) + nn] = (_Float16)(acc[j][v] * inv);
  }
}

// ---------------------------------------------------------------------------
// 5) out = x + W_out @ y + b_out  (512x512 @ 512x4096 + residual)
//    both tiles async-DMA'd to LDS
// ---------------------------------------------------------------------------
__global__ __launch_bounds__(128) void sa2d_outproj(const float* __restrict__ x,
                                                    const _Float16* __restrict__ wo16,
                                                    const float* __restrict__ b_out,
                                                    const _Float16* __restrict__ ybuf,
                                                    float* __restrict__ out) {
  int n = blockIdx.z, ob = blockIdx.y * 64, sb = blockIdx.x * 64;
  int tid = threadIdx.x, lane = tid & 31, wv = tid >> 5;
  __shared__ __align__(16) _Float16 aT[64 * 40];  // [o][c]
  __shared__ __align__(16) _Float16 bT[64 * 40];  // [s][c]
  v8f cacc[2][2] = {};
  int mrow = (wv >> 1) << 5, ncol = (wv & 1) << 5;

  for (int kc = 0; kc < 512; kc += 32) {
    int hd = kc >> 6, dbase = kc & 63;
#pragma unroll
    for (int i = 0; i < 2; ++i) {
      int lin = tid + (i << 7);
      int oI = lin >> 2, ch = lin & 3;
      cp16(aT + oI * 40 + ch * 8, wo16 + (size_t)(ob + oI) * 512 + kc + ch * 8);
    }
#pragma unroll
    for (int i = 0; i < 2; ++i) {  // y tile: 32 c's live inside one head block -> contiguous
      int lin = tid + (i << 7);
      int sI = lin >> 2, ch = lin & 3;
      cp16(bT + sI * 40 + ch * 8,
           ybuf + (((size_t)n * NHD + hd) * SS + sb + sI) * DHD + dbase + ch * 8);
    }
    async_wait();
    __syncthreads();
    v16h A0 = frag_A(aT + mrow * 40, 40, lane);
    v16h A1 = frag_A(aT + (mrow + 16) * 40, 40, lane);
    v16h B0 = frag_B(bT + ncol * 40, 40, lane);
    v16h B1 = frag_B(bT + (ncol + 16) * 40, 40, lane);
    cacc[0][0] = wmma16(A0, B0, cacc[0][0]);
    cacc[0][1] = wmma16(A0, B1, cacc[0][1]);
    cacc[1][0] = wmma16(A1, B0, cacc[1][0]);
    cacc[1][1] = wmma16(A1, B1, cacc[1][1]);
    __syncthreads();
  }

  int half = lane >> 4, nn = lane & 15;
#pragma unroll
  for (int mi = 0; mi < 2; ++mi)
#pragma unroll
    for (int ni = 0; ni < 2; ++ni)
#pragma unroll
      for (int v = 0; v < 8; ++v) {
        int o = ob + mrow + (mi << 4) + v + (half << 3);
        int sIdx = sb + ncol + (ni << 4) + nn;
        size_t idx = ((size_t)n * CC + o) * SS + sIdx;
        out[idx] = x[idx] + cacc[mi][ni][v] + b_out[o];
      }
}

// ---------------------------------------------------------------------------
extern "C" void kernel_launch(void* const* d_in, const int* in_sizes, int n_in,
                              void* d_out, int out_size, void* d_ws, size_t ws_size,
                              hipStream_t stream) {
  const float* x     = (const float*)d_in[0];
  const float* cond  = (const float*)d_in[1];
  const float* W_ada = (const float*)d_in[2];
  const float* b_ada = (const float*)d_in[3];
  const float* W_qkv = (const float*)d_in[4];
  const float* b_qkv = (const float*)d_in[5];
  const float* W_out = (const float*)d_in[6];
  const float* b_out = (const float*)d_in[7];
  float* out = (float*)d_out;

  // workspace layout
  float* stats    = (float*)d_ws;            // 16 f32
  float* partials = stats + 16;              // 1024 f32
  float* ss       = partials + 1024;         // 8192 f32
  _Float16* qkvbuf = (_Float16*)((char*)d_ws + 65536);          // 8*24*4096*64 f16
  _Float16* ybuf   = qkvbuf + (size_t)NB * 24 * SS * DHD;       // 8*8*4096*64 f16
  _Float16* wq16   = ybuf + (size_t)NB * NHD * SS * DHD;        // 1536*512 f16
  _Float16* wo16   = wq16 + (size_t)1536 * 512;                 // 512*512 f16

  sa2d_cvtw<<<1024, 256, 0, stream>>>(W_qkv, W_out, wq16, wo16);
  sa2d_ada <<<NB, 256, 0, stream>>>(cond, W_ada, b_ada, ss);
  sa2d_red1<<<dim3(64, NB), 256, 0, stream>>>(x, partials);
  sa2d_red2<<<NB, 64, 0, stream>>>(partials, stats);
  sa2d_qkv <<<dim3(SS / 64, 1536 / 64, NB), 128, 0, stream>>>(x, wq16, b_qkv, ss, stats, qkvbuf);
  sa2d_flash<<<dim3(SS / 128, NB * NHD), 256, 0, stream>>>(qkvbuf, ybuf);
  sa2d_outproj<<<dim3(SS / 64, CC / 64, NB), 128, 0, stream>>>(x, wo16, b_out, ybuf, out);
}
